// CompressedLinearLayer_15281493639938
// MI455X (gfx1250) — matrix operand order repaired and checked by reading, not compiled, not run on hardware
//
#include <hip/hip_runtime.h>

typedef __bf16        bf16x16 __attribute__((ext_vector_type(16)));
typedef float         f32x8   __attribute__((ext_vector_type(8)));
typedef unsigned int  u32x4   __attribute__((ext_vector_type(4)));
typedef unsigned int  u32x2   __attribute__((ext_vector_type(2)));
typedef int           i32x4   __attribute__((ext_vector_type(4)));

#define D_OUT 4096
#define D_IN  4096
#define RANK  512
#define NTOK  (4 * 2048)   // BATCH * SEQ = 8192 tokens

#define BM 128
#define BN 128
#define BK 32
#define LDSROW 40          // 32 bf16 + 8 pad (80 bytes; rows stay 16B-aligned)

// CDNA5 async global->LDS copy path (ASYNCcnt-tracked), with sync fallback.
#if __has_builtin(__builtin_amdgcn_global_load_async_to_lds_b128) && \
    __has_builtin(__builtin_amdgcn_s_wait_asynccnt)
#define HAVE_ASYNC_LDS 1
#else
#define HAVE_ASYNC_LDS 0
#endif

#if HAVE_ASYNC_LDS
__device__ __forceinline__ void async_copy16(const unsigned short* g, unsigned short* l) {
  __builtin_amdgcn_global_load_async_to_lds_b128(
      (__attribute__((address_space(1))) i32x4*)g,
      (__attribute__((address_space(3))) i32x4*)l, 0, 0);
}
#endif

// ---------- fp32 -> bf16 split helpers (RNE) ----------
__device__ __forceinline__ unsigned short f2bf_rne(float x) {
  unsigned int u = __float_as_uint(x);
  u += 0x7FFFu + ((u >> 16) & 1u);
  return (unsigned short)(u >> 16);
}
__device__ __forceinline__ float bf2f(unsigned short h) {
  return __uint_as_float(((unsigned int)h) << 16);
}
__device__ __forceinline__ void bf16split(float x, unsigned short& h, unsigned short& l) {
  h = f2bf_rne(x);
  l = f2bf_rne(x - bf2f(h));
}

// ---------- prep: elementwise split (A -> Ah, Al), layout preserved ----------
__global__ __launch_bounds__(256) void split_kernel(const float* __restrict__ src,
                                                    unsigned short* __restrict__ dh,
                                                    unsigned short* __restrict__ dl,
                                                    int n) {
  int i = (blockIdx.x * 256 + threadIdx.x) * 4;
  if (i >= n) return;
  float4 v = *(const float4*)(src + i);
  float vv[4] = {v.x, v.y, v.z, v.w};
  unsigned short h[4], l[4];
#pragma unroll
  for (int j = 0; j < 4; ++j) bf16split(vv[j], h[j], l[j]);
  u32x2 ph = { (unsigned)h[0] | ((unsigned)h[1] << 16),
               (unsigned)h[2] | ((unsigned)h[3] << 16) };
  u32x2 pl = { (unsigned)l[0] | ((unsigned)l[1] << 16),
               (unsigned)l[2] | ((unsigned)l[3] << 16) };
  *(u32x2*)(dh + i) = ph;
  *(u32x2*)(dl + i) = pl;
}

// ---------- prep: transpose + split B [D_IN,RANK] -> Bt hi/lo [RANK,D_IN] ----------
__global__ __launch_bounds__(256) void transpose_split_kernel(const float* __restrict__ B,
                                                              unsigned short* __restrict__ th,
                                                              unsigned short* __restrict__ tl) {
  __shared__ float tile[32][33];
  const int r0 = blockIdx.x * 32;   // rank
  const int d0 = blockIdx.y * 32;   // d_in
  const int tx = threadIdx.x, ty = threadIdx.y;   // 32 x 8
#pragma unroll
  for (int i = 0; i < 32; i += 8)
    tile[ty + i][tx] = B[(size_t)(d0 + ty + i) * RANK + (r0 + tx)];
  __syncthreads();
#pragma unroll
  for (int i = 0; i < 32; i += 8) {
    float v = tile[tx][ty + i];     // = B[d0+tx][r0+ty+i]
    unsigned short h, l;
    bf16split(v, h, l);
    size_t o = (size_t)(r0 + ty + i) * D_IN + (d0 + tx);
    th[o] = h;
    tl[o] = l;
  }
}

// ---------- NT GEMM, 2-term bf16 split, fp32 accumulate, double-buffered LDS ----
// C[M,N] = L[M,K] * R[N,K]^T (+ bias). R pre-split bf16 hi/lo.
// L_IS_F32: L is fp32, split on the fly; else L pre-split bf16 hi/lo.
// SPLIT_OUT: write result as bf16 hi/lo (Ch/Cl); else fp32 C (+bias).
template <bool L_IS_F32, bool SPLIT_OUT>
__global__ __launch_bounds__(256) void gemm_nt(
    const float* __restrict__ Lf,
    const unsigned short* __restrict__ Lh, const unsigned short* __restrict__ Ll,
    const unsigned short* __restrict__ Rh, const unsigned short* __restrict__ Rl,
    const float* __restrict__ bias,
    float* __restrict__ C,
    unsigned short* __restrict__ Ch, unsigned short* __restrict__ Cl,
    int M, int N, int K) {
  // [buffer][array: 0=Lh 1=Ll 2=Rh 3=Rl][tile]
  __shared__ unsigned short lds[2][4][BM * LDSROW];

  const int tid   = threadIdx.x;
  const int lane  = tid & 31;
  const int wave  = tid >> 5;
  const int waveM = wave >> 2;   // 0..1  (2 waves along M)
  const int waveN = wave & 3;    // 0..3  (4 waves along N)
  const int lm    = lane & 15;
  const int lh    = lane >> 4;   // lane-half selects K-chunk

  const int m0 = blockIdx.y * BM;
  const int n0 = blockIdx.x * BN;

  f32x8 acc[4][2] = {};

  auto load_tiles = [&](int kt, int buf) {
    if constexpr (L_IS_F32) {
      // 128x32 fp32 -> hi/lo bf16 : 4 float4 chunks per thread (must round-trip VGPRs)
#pragma unroll
      for (int it = 0; it < 4; ++it) {
        int flat = tid + it * 256;
        int row  = flat >> 3;
        int c4   = (flat & 7) << 2;
        float4 v = *(const float4*)(Lf + (size_t)(m0 + row) * K + kt + c4);
        float vv[4] = {v.x, v.y, v.z, v.w};
        unsigned short h[4], l[4];
#pragma unroll
        for (int j = 0; j < 4; ++j) bf16split(vv[j], h[j], l[j]);
        u32x2 ph = { (unsigned)h[0] | ((unsigned)h[1] << 16),
                     (unsigned)h[2] | ((unsigned)h[3] << 16) };
        u32x2 pl = { (unsigned)l[0] | ((unsigned)l[1] << 16),
                     (unsigned)l[2] | ((unsigned)l[3] << 16) };
        *(u32x2*)(&lds[buf][0][row * LDSROW + c4]) = ph;
        *(u32x2*)(&lds[buf][1][row * LDSROW + c4]) = pl;
      }
    } else {
      int row = tid >> 1;
      int ch  = (tid & 1) << 3;
      size_t go = (size_t)(m0 + row) * K + kt + ch;
      size_t lo = (size_t)row * LDSROW + ch;
#if HAVE_ASYNC_LDS
      async_copy16(Lh + go, &lds[buf][0][lo]);
      async_copy16(Ll + go, &lds[buf][1][lo]);
#else
      *(u32x4*)(&lds[buf][0][lo]) = *(const u32x4*)(Lh + go);
      *(u32x4*)(&lds[buf][1][lo]) = *(const u32x4*)(Ll + go);
#endif
    }
    {
      int row = tid >> 1;
      int ch  = (tid & 1) << 3;
      size_t go = (size_t)(n0 + row) * K + kt + ch;
      size_t lo = (size_t)row * LDSROW + ch;
#if HAVE_ASYNC_LDS
      async_copy16(Rh + go, &lds[buf][2][lo]);
      async_copy16(Rl + go, &lds[buf][3][lo]);
#else
      *(u32x4*)(&lds[buf][2][lo]) = *(const u32x4*)(Rh + go);
      *(u32x4*)(&lds[buf][3][lo]) = *(const u32x4*)(Rl + go);
#endif
    }
  };

  load_tiles(0, 0);
#if HAVE_ASYNC_LDS
  __builtin_amdgcn_s_wait_asynccnt(0);
#endif
  __syncthreads();

  const int nk = K / BK;
  for (int ik = 0; ik < nk; ++ik) {
    const int buf = ik & 1;
    if (ik + 1 < nk) load_tiles((ik + 1) * BK, buf ^ 1);   // fill other buffer
    if (ik + 2 < nk) {                                     // prefetch tile k+2
      int kp = (ik + 2) * BK;
      if constexpr (L_IS_F32)
        __builtin_prefetch(Lf + (size_t)(m0 + (tid >> 3)) * K + kp, 0, 0);
      else
        __builtin_prefetch(Lh + (size_t)(m0 + (tid >> 1)) * K + kp, 0, 0);
      __builtin_prefetch(Rh + (size_t)(n0 + (tid >> 1)) * K + kp, 0, 0);
      __builtin_prefetch(Rl + (size_t)(n0 + (tid >> 1)) * K + kp, 0, 0);
    }

    // ---- B operands: elems 0..7 = K[8*lh ..], elems 8..15 = K[16+8*lh ..] ----
    bf16x16 bh[2], bl[2];
#pragma unroll
    for (int ni = 0; ni < 2; ++ni) {
      int row = waveN * 32 + ni * 16 + lm;
      const unsigned short* ph = &lds[buf][2][row * LDSROW + lh * 8];
      const unsigned short* pl = &lds[buf][3][row * LDSROW + lh * 8];
      union { u32x4 q[2]; bf16x16 v; } t;
      t.q[0] = *(const u32x4*)(ph); t.q[1] = *(const u32x4*)(ph + 16); bh[ni] = t.v;
      t.q[0] = *(const u32x4*)(pl); t.q[1] = *(const u32x4*)(pl + 16); bl[ni] = t.v;
    }

    // ---- A operands: elems 0..7 = K[8*lh ..], elems 8..15 = K[16+8*lh ..] ----
#pragma unroll
    for (int mi = 0; mi < 4; ++mi) {
      int row = waveM * 64 + mi * 16 + lm;
      const unsigned short* ph = &lds[buf][0][row * LDSROW];
      const unsigned short* pl = &lds[buf][1][row * LDSROW];
      union { u32x4 q[2]; bf16x16 v; } ta, tb;
      ta.q[0] = *(const u32x4*)(ph + lh * 8);
      ta.q[1] = *(const u32x4*)(ph + 16 + lh * 8);
      tb.q[0] = *(const u32x4*)(pl + lh * 8);
      tb.q[1] = *(const u32x4*)(pl + 16 + lh * 8);
      bf16x16 ah = ta.v, al = tb.v;
#pragma unroll
      for (int ni = 0; ni < 2; ++ni) {
        acc[mi][ni] = __builtin_amdgcn_wmma_f32_16x16x32_bf16(
            false, ah, false, bh[ni], (short)0, acc[mi][ni], false, false);
        acc[mi][ni] = __builtin_amdgcn_wmma_f32_16x16x32_bf16(
            false, al, false, bh[ni], (short)0, acc[mi][ni], false, false);
        acc[mi][ni] = __builtin_amdgcn_wmma_f32_16x16x32_bf16(
            false, ah, false, bl[ni], (short)0, acc[mi][ni], false, false);
      }
    }
#if HAVE_ASYNC_LDS
    __builtin_amdgcn_s_wait_asynccnt(0);   // my async fills of the other buffer done
#endif
    __syncthreads();                        // everyone's fills done
  }

  // ---- epilogue: acc elem j -> (m = 8*lh + j, n = lm) within each 16x16 tile ----
#pragma unroll
  for (int mi = 0; mi < 4; ++mi) {
#pragma unroll
    for (int ni = 0; ni < 2; ++ni) {
      int col = n0 + waveN * 32 + ni * 16 + lm;
      int rowbase = m0 + waveM * 64 + mi * 16 + lh * 8;
      if constexpr (SPLIT_OUT) {
#pragma unroll
        for (int j = 0; j < 8; ++j) {
          unsigned short h, l;
          bf16split(acc[mi][ni][j], h, l);
          size_t o = (size_t)(rowbase + j) * N + col;
          Ch[o] = h;
          Cl[o] = l;
        }
      } else {
        float bv = bias ? bias[col] : 0.0f;
#pragma unroll
        for (int j = 0; j < 8; ++j)
          C[(size_t)(rowbase + j) * N + col] = acc[mi][ni][j] + bv;
      }
    }
  }
}

extern "C" void kernel_launch(void* const* d_in, const int* in_sizes, int n_in,
                              void* d_out, int out_size, void* d_ws, size_t ws_size,
                              hipStream_t stream) {
  const float* x    = (const float*)d_in[0];   // [4, 2048, 4096]
  const float* A    = (const float*)d_in[1];   // [4096, 512]
  const float* B    = (const float*)d_in[2];   // [4096, 512]
  const float* bias = (const float*)d_in[3];   // [4096]
  float* out = (float*)d_out;                  // [8192, 4096]

  // workspace layout (32 MB total)
  unsigned short* Ah  = (unsigned short*)d_ws;
  unsigned short* Al  = Ah  + (size_t)D_OUT * RANK;
  unsigned short* Bth = Al  + (size_t)D_OUT * RANK;
  unsigned short* Btl = Bth + (size_t)RANK * D_IN;
  unsigned short* Th  = Btl + (size_t)RANK * D_IN;   // [8192, 512] bf16 hi
  unsigned short* Tl  = Th  + (size_t)NTOK * RANK;   // [8192, 512] bf16 lo

  // prep: split A (already NT-ready), transpose+split B
  split_kernel<<<(D_OUT * RANK / 4 + 255) / 256, 256, 0, stream>>>(A, Ah, Al, D_OUT * RANK);
  transpose_split_kernel<<<dim3(RANK / 32, D_IN / 32), dim3(32, 8), 0, stream>>>(B, Bth, Btl);

  // stage 1: T = x @ B  (x fp32 split on the fly; T written pre-split bf16 hi/lo)
  gemm_nt<true, true><<<dim3(RANK / BN, NTOK / BM), 256, 0, stream>>>(
      x, nullptr, nullptr, Bth, Btl, nullptr, nullptr, Th, Tl, NTOK, RANK, D_IN);

  // stage 2: out = T @ A^T + bias  (all operands pre-split bf16 -> zero conversion VALU)
  gemm_nt<false, false><<<dim3(D_OUT / BN, NTOK / BM), 256, 0, stream>>>(
      nullptr, Th, Tl, Ah, Al, bias, out, nullptr, nullptr, NTOK, D_OUT, RANK);
}
